// CostVolume_42219528520127
// MI455X (gfx1250) — matrix operand order repaired and checked by reading, not compiled, not run on hardware
//
#include <hip/hip_runtime.h>
#include <hip/hip_bf16.h>

typedef __attribute__((ext_vector_type(16))) _Float16 v16h;
typedef __attribute__((ext_vector_type(8)))  float    v8f;

#define LRELU_SLOPE 0.01f
#define BN_EPS_F    1e-5f

struct WB   { const _Float16* w; const float* b; };
struct W1set { WB l0, l1, l2, pienc, m2a, m2b; };
struct W2set { WB pcenc, l0, l1; };

// ---------------------------------------------------------------------------
// WMMA helpers (wave32, v_wmma_f32_16x16x32_f16)
// A layout (16x32 f16): lane m=lane&15, h=lane>>4;
//   halfs[2j+s]  -> k = 8h+2j+s        (j=0..3)
//   halfs[8+2j+s]-> k = 16+8h+2j+s     (j=0..3)
// B layout (32x16 f16): lane n=lane&15, h=lane>>4; same k order as A
// C/D layout (16x16 f32): vgpr j, lane l: row = 8*(l>>4)+j, col = l&15
// ---------------------------------------------------------------------------
__device__ __forceinline__ v16h load_a_frag(const _Float16* X, int row_base,
                                            int stride, int kt) {
  const int lane = threadIdx.x & 31;
  const int m = lane & 15, h = lane >> 4;
  const _Float16* p = X + (row_base + m) * stride + kt * 32;
  v16h a;
#pragma unroll
  for (int j = 0; j < 4; ++j) {
    a[2 * j]     = p[8 * h + 2 * j];
    a[2 * j + 1] = p[8 * h + 2 * j + 1];
    a[8 + 2 * j] = p[16 + 8 * h + 2 * j];
    a[9 + 2 * j] = p[16 + 8 * h + 2 * j + 1];
  }
  return a;
}

// 64-row (4 M-tile) GEMM: X[64 x 32*KT] (lds,f16) * Wfrag -> Y[64 x 16*NT]
// with folded-BN bias + LeakyReLU. Runs on a 128-thread block (4 waves).
template <int KT, int NT>
__device__ __forceinline__ void gemm_f16(const _Float16* X, int xs,
                                         const _Float16* Wf, const float* bias,
                                         _Float16* Y, int ys) {
  const int lane = threadIdx.x & 31;
  const int row_base = (threadIdx.x >> 5) * 16;   // wave -> M tile
  __builtin_prefetch(Wf, 0, 1);                   // global_prefetch_b8
  v16h a[KT];
#pragma unroll
  for (int kt = 0; kt < KT; ++kt) a[kt] = load_a_frag(X, row_base, xs, kt);
  const int n_lo = lane & 15;
  const int mh = (lane >> 4) * 8;
  const v16h* wf = reinterpret_cast<const v16h*>(Wf);
#pragma unroll
  for (int nt = 0; nt < NT; ++nt) {
    v8f acc = {};
#pragma unroll
    for (int kt = 0; kt < KT; ++kt) {
      v16h b = wf[(nt * KT + kt) * 32 + lane];
      acc = __builtin_amdgcn_wmma_f32_16x16x32_f16(false, a[kt], false, b,
                                                   (short)0, acc, false, false);
    }
    const int n = nt * 16 + n_lo;
    const float bv = bias[n];
#pragma unroll
    for (int j = 0; j < 8; ++j) {
      float y = acc[j] + bv;
      y = (y >= 0.f) ? y : LRELU_SLOPE * y;
      Y[(row_base + mh + j) * ys + n] = (_Float16)y;
    }
  }
}

// ---------------------------------------------------------------------------
// Weight prep: fold eval-BN into W/bias, emit f16 B-fragments in ISA layout.
// ---------------------------------------------------------------------------
__global__ void prep_weights_kernel(const float* __restrict__ W,
                                    const float* __restrict__ gamma,
                                    const float* __restrict__ beta,
                                    const float* __restrict__ mean,
                                    const float* __restrict__ var,
                                    int Cin, int Cout, int Kt, int Nt,
                                    _Float16* __restrict__ frag,
                                    float* __restrict__ bias) {
  const int t = blockIdx.x * blockDim.x + threadIdx.x;
  const int total = Nt * Kt * 512;
  if (t < total) {
    const int e = t & 15;
    const int lane = (t >> 4) & 31;
    const int tile = t >> 9;
    const int kt = tile % Kt, nt = tile / Kt;
    const int half = lane >> 4, nl = lane & 15;
    const int j = e >> 1, s = e & 1;
    const int k = kt * 32 + 16 * (j >> 2) + 8 * half + 2 * (j & 3) + s;
    const int n = nt * 16 + nl;
    float v = 0.f;
    if (k < Cin && n < Cout)
      v = W[n * Cin + k] * (gamma[n] * rsqrtf(var[n] + BN_EPS_F));
    frag[tile * 512 + lane * 16 + e] = (_Float16)v;
  }
  if (t < Cout)
    bias[t] = beta[t] - mean[t] * (gamma[t] * rsqrtf(var[t] + BN_EPS_F));
}

// ---------------------------------------------------------------------------
// Per-point prep: scaled coords + validity mask (on unscaled coords).
// ---------------------------------------------------------------------------
__global__ void prep_points_kernel(const float* __restrict__ wxyz,
                                   const float* __restrict__ lz,
                                   float* __restrict__ wxyz_s,
                                   float* __restrict__ valid, int N) {
  const int t = blockIdx.x * blockDim.x + threadIdx.x;
  if (t >= N) return;
  const float x = wxyz[t * 3], y = wxyz[t * 3 + 1], z = wxyz[t * 3 + 2];
  valid[t] = (x * x + y * y + z * z >= 1e-10f) ? 1.f : 0.f;
  const float s = lz[t];
  wxyz_s[t * 3] = x * s; wxyz_s[t * 3 + 1] = y * s; wxyz_s[t * 3 + 2] = z * s;
}

// ---------------------------------------------------------------------------
// Brute-force KNN, one wave32 per query; distances cached in LDS, then K
// rounds of lexicographic (dist,idx) min extraction (matches top_k tie order).
// ---------------------------------------------------------------------------
__launch_bounds__(64, 1)
__global__ void knn_kernel(const float* __restrict__ qpts,
                           const float* __restrict__ cpts,
                           const float* __restrict__ penalty,  // may be null
                           const int* __restrict__ binfo,
                           const int* __restrict__ bszp,
                           int N, int Mtot, int K,
                           int* __restrict__ outidx) {
  __shared__ float sdist[2 * 4096];
  const int wave = threadIdx.x >> 5, lane = threadIdx.x & 31;
  const int q = blockIdx.x * 2 + wave;
  const int B = bszp[0];
  const int m = Mtot / B;
  const int b = binfo[q];
  const float qx = qpts[q * 3], qy = qpts[q * 3 + 1], qz = qpts[q * 3 + 2];
  float* dl = sdist + wave * 4096;
  const bool pen = (penalty != nullptr);
  for (int c = lane; c < m; c += 32) {
    const float* p = cpts + (size_t)(b * m + c) * 3;
    const float dx = qx - p[0], dy = qy - p[1], dz = qz - p[2];
    float d = dx * dx + dy * dy + dz * dz;
    if (pen) d += (1.f - penalty[b * m + c]) * 1e10f;
    dl[c] = d;
  }
  __syncthreads();
  unsigned long long prev = 0ull;
  for (int k = 0; k < K; ++k) {
    unsigned long long best = ~0ull;
    for (int c = lane; c < m; c += 32) {
      const unsigned long long key =
          ((((unsigned long long)__float_as_uint(dl[c])) << 32) |
           (unsigned)c) + 1ull;
      if (key > prev && key < best) best = key;
    }
#pragma unroll
    for (int off = 16; off > 0; off >>= 1) {
      unsigned lo = (unsigned)__shfl_xor((int)(unsigned)best, off, 32);
      unsigned hi = (unsigned)__shfl_xor((int)(unsigned)(best >> 32), off, 32);
      const unsigned long long o = (((unsigned long long)hi) << 32) | lo;
      if (o < best) best = o;
    }
    prev = best;
    if (lane == 0)
      outidx[q * K + k] = b * m + (int)((best - 1ull) & 0xffffffffull);
  }
}

// ---------------------------------------------------------------------------
// Stage 1: per block = 2 queries x 32 neighbors = 64 rows, all layers fused.
// ---------------------------------------------------------------------------
__launch_bounds__(128, 1)
__global__ void stage1_kernel(const float* __restrict__ wxyz_s,
                              const float* __restrict__ wpts,
                              const float* __restrict__ f2_xyz,
                              const float* __restrict__ f2_pts,
                              const int* __restrict__ knnq,
                              float* __restrict__ pi_agg, W1set W) {
  __shared__ _Float16 sFeat[64 * 96];   // mlp1 input (70 used) / pi_enc input
  __shared__ _Float16 sY[64 * 128];     // mlp1 L0 out, later mlp2 L0 out
  __shared__ _Float16 sD[64 * 64];      // mlp1 L1 out
  __shared__ _Float16 sC[64 * 128];     // [pi_enc(64) | pi_feat1_new(64)]
  __shared__ _Float16 sL[64 * 64];      // mlp2 logits
  __shared__ float qn[2 * 64];
  __shared__ float qx[2 * 3];

  const int t = threadIdx.x;
  const int q0 = blockIdx.x * 2;

  if (t < 2) {  // per-query scaled xyz + normalized warped_points
    const int g = q0 + t;
    for (int i = 0; i < 3; ++i) qx[t * 3 + i] = wxyz_s[g * 3 + i];
    const float* wp = wpts + (size_t)g * 64;
    float mu = 0.f;
    for (int c = 0; c < 64; ++c) mu += wp[c];
    mu *= (1.f / 64.f);
    float s2 = 0.f;
    for (int c = 0; c < 64; ++c) { const float d = wp[c] - mu; s2 += d * d; }
    const float inv = 1.f / fmaxf(sqrtf(s2 * (1.f / 63.f)), 1e-12f);
    for (int c = 0; c < 64; ++c) qn[t * 64 + c] = (wp[c] - mu) * inv;
  }
  __syncthreads();

  if (t < 64) {  // one row per thread: gather + normalize + feature build
    const int qi = t >> 5;
    const int g = q0 + qi;
    const int idx = knnq[g * 32 + (t & 31)];
    _Float16* fr = sFeat + t * 96;
    fr[0] = (_Float16)qx[qi * 3 + 0];
    fr[1] = (_Float16)qx[qi * 3 + 1];
    fr[2] = (_Float16)qx[qi * 3 + 2];
    const float* gx = f2_xyz + (size_t)idx * 3;
    fr[3] = (_Float16)gx[0]; fr[4] = (_Float16)gx[1]; fr[5] = (_Float16)gx[2];
    const float* gp = f2_pts + (size_t)idx * 64;
    float mu = 0.f;
    for (int c = 0; c < 64; ++c) mu += gp[c];
    mu *= (1.f / 64.f);
    float s2 = 0.f;
    for (int c = 0; c < 64; ++c) { const float d = gp[c] - mu; s2 += d * d; }
    const float inv = 1.f / fmaxf(sqrtf(s2 * (1.f / 63.f)), 1e-12f);
    for (int c = 0; c < 64; ++c)
      fr[6 + c] = (_Float16)(qn[qi * 64 + c] * (gp[c] - mu) * inv);
    for (int c = 70; c < 96; ++c) fr[c] = (_Float16)0.f;
  }
  __syncthreads();

  gemm_f16<3, 8>(sFeat, 96, W.l0.w, W.l0.b, sY, 128);     // 70->128
  __syncthreads();
  gemm_f16<4, 4>(sY, 128, W.l1.w, W.l1.b, sD, 64);        // 128->64
  __syncthreads();
  gemm_f16<2, 4>(sD, 64, W.l2.w, W.l2.b, sC + 64, 128);   // 64->64 (concat hi)
  gemm_f16<1, 4>(sFeat, 96, W.pienc.w, W.pienc.b, sC, 128); // pi_enc 6->64
  __syncthreads();
  gemm_f16<4, 8>(sC, 128, W.m2a.w, W.m2a.b, sY, 128);     // 128->128
  __syncthreads();
  gemm_f16<4, 4>(sY, 128, W.m2b.w, W.m2b.b, sL, 64);      // 128->64 logits
  __syncthreads();

  {  // softmax over 32 neighbors, weighted sum of pi_feat1_new
    const int qi = t >> 6, c = t & 63;
    const _Float16* lrow = sL + qi * 32 * 64 + c;
    float mx = -1e30f;
    for (int r = 0; r < 32; ++r) mx = fmaxf(mx, (float)lrow[r * 64]);
    float se = 0.f, num = 0.f;
    const _Float16* frow = sC + qi * 32 * 128 + 64 + c;
    for (int r = 0; r < 32; ++r) {
      const float e = __expf((float)lrow[r * 64] - mx);
      se += e;
      num += e * (float)frow[r * 128];
    }
    pi_agg[(size_t)(q0 + qi) * 64 + c] = num / se;
  }
}

// ---------------------------------------------------------------------------
// Stage 2: per block = 4 queries x 16 neighbors = 64 rows, fused.
// ---------------------------------------------------------------------------
__launch_bounds__(128, 1)
__global__ void stage2_kernel(const float* __restrict__ wxyz_s,
                              const float* __restrict__ wpts,
                              const float* __restrict__ pi_agg,
                              const int* __restrict__ knns,
                              float* __restrict__ out, W2set W) {
  __shared__ _Float16 sFeat[64 * 32];   // pc_xyz_diff_concat (10 used)
  __shared__ _Float16 sC[64 * 192];     // [pc_enc(64)|wpts(64)|grouped(64)]
  __shared__ _Float16 sY[64 * 128];
  __shared__ _Float16 sL[64 * 64];
  const int t = threadIdx.x;
  const int q0 = blockIdx.x * 4;

  if (t < 64) {  // xyz features, one row per thread
    const int qi = t >> 4;
    const int g = q0 + qi;
    const int idx = knns[g * 16 + (t & 15)];
    const float px0 = wxyz_s[g * 3], px1 = wxyz_s[g * 3 + 1],
                px2 = wxyz_s[g * 3 + 2];
    const float gx0 = wxyz_s[idx * 3], gx1 = wxyz_s[idx * 3 + 1],
                gx2 = wxyz_s[idx * 3 + 2];
    const float d0 = gx0 - px0, d1 = gx1 - px1, d2 = gx2 - px2;
    const float eu = sqrtf(d0 * d0 + d1 * d1 + d2 * d2 + 1e-20f);
    _Float16* fr = sFeat + t * 32;
    fr[0] = (_Float16)px0; fr[1] = (_Float16)px1; fr[2] = (_Float16)px2;
    fr[3] = (_Float16)gx0; fr[4] = (_Float16)gx1; fr[5] = (_Float16)gx2;
    fr[6] = (_Float16)d0;  fr[7] = (_Float16)d1;  fr[8] = (_Float16)d2;
    fr[9] = (_Float16)eu;
    for (int c = 10; c < 32; ++c) fr[c] = (_Float16)0.f;
  }
  {  // concat mid/high thirds: row r, 32-channel strip h
    const int r = t & 63, h = t >> 6;
    const int qi = r >> 4;
    const int g = q0 + qi;
    const int idx = knns[g * 16 + (r & 15)];
    const float* wp = wpts + (size_t)g * 64 + h * 32;
    const float* pg = pi_agg + (size_t)idx * 64 + h * 32;
    _Float16* row = sC + r * 192;
    for (int c = 0; c < 32; ++c) {
      row[64 + h * 32 + c] = (_Float16)wp[c];
      row[128 + h * 32 + c] = (_Float16)pg[c];
    }
  }
  __syncthreads();

  gemm_f16<1, 4>(sFeat, 32, W.pcenc.w, W.pcenc.b, sC, 192); // pc_enc 10->64
  __syncthreads();
  gemm_f16<6, 8>(sC, 192, W.l0.w, W.l0.b, sY, 128);         // 192->128
  __syncthreads();
  gemm_f16<4, 4>(sY, 128, W.l1.w, W.l1.b, sL, 64);          // 128->64 logits
  __syncthreads();

  for (int p = t; p < 256; p += 128) {  // softmax over 16 neighbors
    const int qi = p >> 6, c = p & 63;
    const _Float16* lrow = sL + qi * 16 * 64 + c;
    float mx = -1e30f;
    for (int r = 0; r < 16; ++r) mx = fmaxf(mx, (float)lrow[r * 64]);
    float se = 0.f, num = 0.f;
    const _Float16* frow = sC + qi * 16 * 192 + 128 + c;
    for (int r = 0; r < 16; ++r) {
      const float e = __expf((float)lrow[r * 64] - mx);
      se += e;
      num += e * (float)frow[r * 192];
    }
    out[(size_t)(q0 + qi) * 64 + c] = num / se;  // nc2bnc ind == identity
  }
}

// ---------------------------------------------------------------------------
extern "C" void kernel_launch(void* const* d_in, const int* in_sizes, int n_in,
                              void* d_out, int out_size, void* d_ws,
                              size_t ws_size, hipStream_t stream) {
  const float* wxyz  = (const float*)d_in[0];
  const float* wpts  = (const float*)d_in[1];
  const int*   binfo = (const int*)d_in[2];
  const int*   bszp  = (const int*)d_in[3];
  const float* f2xyz = (const float*)d_in[4];
  const float* f2pts = (const float*)d_in[5];
  const float* lz    = (const float*)d_in[6];
  const int N = in_sizes[0] / 3;
  const int Mtot = in_sizes[4] / 3;

  char* ws = (char*)d_ws;
  size_t off = 0;
  auto alloc = [&](size_t bytes) -> void* {
    off = (off + 255) & ~(size_t)255;
    void* p = ws + off;
    off += bytes;
    return p;
  };
  float* wxyz_s = (float*)alloc((size_t)N * 3 * 4);
  float* valid  = (float*)alloc((size_t)N * 4);
  int*   knnq   = (int*)alloc((size_t)N * 32 * 4);
  int*   knns   = (int*)alloc((size_t)N * 16 * 4);
  float* piagg  = (float*)alloc((size_t)N * 64 * 4);

  // layer table: {d_in index of W, Cin, Cout, Ktiles, Ntiles}
  struct LD { int i, cin, cout, kt, nt; };
  const LD defs[9] = {
      {7, 70, 128, 3, 8},  {12, 128, 64, 4, 4}, {17, 64, 64, 2, 4},
      {22, 6, 64, 1, 4},   {27, 128, 128, 4, 8}, {32, 128, 64, 4, 4},
      {37, 10, 64, 1, 4},  {42, 192, 128, 6, 8}, {47, 128, 64, 4, 4}};
  WB wb[9];
  for (int l = 0; l < 9; ++l) {
    const LD& d = defs[l];
    const int elems = d.nt * d.kt * 512;
    _Float16* fr = (_Float16*)alloc((size_t)elems * 2);
    float* bi = (float*)alloc((size_t)d.cout * 4);
    wb[l].w = fr;
    wb[l].b = bi;
    prep_weights_kernel<<<(elems + 255) / 256, 256, 0, stream>>>(
        (const float*)d_in[d.i], (const float*)d_in[d.i + 1],
        (const float*)d_in[d.i + 2], (const float*)d_in[d.i + 3],
        (const float*)d_in[d.i + 4], d.cin, d.cout, d.kt, d.nt, fr, bi);
  }

  prep_points_kernel<<<(N + 255) / 256, 256, 0, stream>>>(wxyz, lz, wxyz_s,
                                                          valid, N);

  // KNN 1: unscaled lidar points vs image points, top-32
  knn_kernel<<<N / 2, 64, 0, stream>>>(wxyz, f2xyz, (const float*)nullptr,
                                       binfo, bszp, N, Mtot, 32, knnq);
  // KNN 2: scaled self-grouping with validity penalty, top-16
  knn_kernel<<<N / 2, 64, 0, stream>>>(wxyz_s, wxyz_s, valid, binfo, bszp, N,
                                       N, 16, knns);

  W1set w1 = {wb[0], wb[1], wb[2], wb[3], wb[4], wb[5]};
  stage1_kernel<<<N / 2, 128, 0, stream>>>(wxyz_s, wpts, f2xyz, f2pts, knnq,
                                           piagg, w1);

  W2set w2 = {wb[6], wb[7], wb[8]};
  stage2_kernel<<<N / 4, 128, 0, stream>>>(wxyz_s, wpts, piagg, knns,
                                           (float*)d_out, w2);
}
